// SimpleConvolutionalGNN_2035814498858
// MI455X (gfx1250) — compile-verified, matrix operand order checked by reading, not run on hardware
//
#include <hip/hip_runtime.h>
#include <hip/hip_bf16.h>

typedef __bf16 bf16;
typedef __attribute__((ext_vector_type(16))) __bf16 v16bf;
typedef __attribute__((ext_vector_type(8)))  __bf16 v8bf;
typedef __attribute__((ext_vector_type(8)))  float  v8f;

#define CH   128   // IN == H == 128
#define RPB  64    // rows per GEMM block
#define LROW 136   // padded LDS row length (272 B = 68 dwords -> 4-bank skew per row)

// Load one 16-element bf16 WMMA fragment = two contiguous 16-byte runs (2x ds_load_b128).
__device__ inline v16bf load_frag(const bf16* base) {
    v8bf lo = *(const v8bf*)(base);        // elements 0..7   (k = base..base+7)
    v8bf hi = *(const v8bf*)(base + 16);   // elements 8..15  (k = base+16..base+23)
    return __builtin_shufflevector(lo, hi, 0, 1, 2, 3, 4, 5, 6, 7,
                                           8, 9, 10, 11, 12, 13, 14, 15);
}

// ---------------- degree / normalization ----------------
__global__ void k_init_deg(float* deg, int N) {
    int i = blockIdx.x * blockDim.x + threadIdx.x;
    if (i < N) deg[i] = 1.0f;                       // self-loop
}

__global__ void k_edge_deg(const long long* __restrict__ ei, float* deg, int E) {
    int e = blockIdx.x * blockDim.x + threadIdx.x;
    if (e < E) atomicAdd(&deg[(int)ei[(long long)E + e]], 1.0f);  // dst row
}

__global__ void k_rsqrt(float* d, int N) {
    int i = blockIdx.x * blockDim.x + threadIdx.x;
    if (i < N) d[i] = rsqrtf(d[i]);
}

// ---------------- WMMA GEMM: Hout[N,128] = (relu?)(A[N,128]) @ W[128,128] ----------------
// 256 threads = 8 waves; wave w owns output columns [16w, 16w+16); block owns 64 rows.
__global__ __launch_bounds__(256)
void k_gemm_wmma(const float* __restrict__ A, const float* __restrict__ W,
                 float* __restrict__ Hout, int N, int relu_in) {
    __shared__ bf16 sA[RPB * LROW];   // 17.4 KB  A rows (bf16), padded
    __shared__ bf16 sW[CH * LROW];    // 34.8 KB  W transposed: sW[n*LROW + k]

    const int tid  = threadIdx.x;
    const int row0 = blockIdx.x * RPB;

    // stage W transposed (coalesced global read, skewed LDS write)
    for (int i = tid; i < CH * CH; i += 256) {
        int k = i >> 7, n = i & (CH - 1);
        sW[n * LROW + k] = (bf16)W[i];
    }
    // stage A rows, fp32 -> bf16, fused ReLU of previous layer
    for (int i = tid; i < RPB * CH; i += 256) {
        int r = i >> 7, c = i & (CH - 1);
        int row = row0 + r;
        float v = (row < N) ? A[row * CH + c] : 0.0f;
        if (relu_in) v = fmaxf(v, 0.0f);
        sA[r * LROW + c] = (bf16)v;
    }
    __syncthreads();

    const int wave = tid >> 5, lane = tid & 31;
    const int m    = lane & 15;        // row-in-tile / col-in-tile
    const int half = lane >> 4;
    const int ncol = wave * 16 + m;    // this wave's output column

    // B fragments: register-resident, reused across all 4 row sub-tiles
    v16bf bfrag[4];
    #pragma unroll
    for (int s = 0; s < 4; ++s)
        bfrag[s] = load_frag(&sW[ncol * LROW + s * 32 + half * 8]);

    const bool full = (row0 + RPB) <= N;   // block-uniform: only last block is ragged

    #pragma unroll
    for (int rt = 0; rt < 4; ++rt) {   // four 16-row sub-tiles
        v8f acc = {};
        #pragma unroll
        for (int s = 0; s < 4; ++s) {  // K = 128 in 4 steps of 32
            v16bf afrag = load_frag(&sA[(rt * 16 + m) * LROW + s * 32 + half * 8]);
            acc = __builtin_amdgcn_wmma_f32_16x16x32_bf16(
                false, afrag, false, bfrag[s], (short)0, acc, false, false);
        }
        // C/D layout: VGPR r -> M = r (lanes 0-15) / r+8 (lanes 16-31), N = lane%16
        float* base = Hout + (row0 + rt * 16 + half * 8) * CH + ncol;
        if (full) {
            #pragma unroll
            for (int r = 0; r < 8; ++r) base[r * CH] = acc[r];
        } else {
            #pragma unroll
            for (int r = 0; r < 8; ++r) {
                int row = row0 + rt * 16 + r + half * 8;
                if (row < N) Hout[row * CH + ncol] = acc[r];
            }
        }
    }
}

// ---------------- out = h * self_norm + bias (float4 per thread) ----------------
__global__ void k_init_out(const float* __restrict__ h, const float* __restrict__ dis,
                           const float* __restrict__ bias, float* __restrict__ out, int N) {
    int i = blockIdx.x * blockDim.x + threadIdx.x;     // over N*32 float4 groups
    if (i >= N * 32) return;
    int n = i >> 5, q = i & 31;
    float d  = dis[n];
    float sn = d * d;
    float4 h4 = ((const float4*)(h + (long long)n * CH))[q];
    float4 b4 = ((const float4*)bias)[q];
    float4 o;
    o.x = h4.x * sn + b4.x;
    o.y = h4.y * sn + b4.y;
    o.z = h4.z * sn + b4.z;
    o.w = h4.w * sn + b4.w;
    ((float4*)(out + (long long)n * CH))[q] = o;
}

// ---------------- edge scatter: out[dst] += norm * h[src]  (one wave per edge) ----------------
__global__ __launch_bounds__(256)
void k_edge_scatter(const long long* __restrict__ ei, const float* __restrict__ dis,
                    const float* __restrict__ h, float* out, int E) {
    int wave = threadIdx.x >> 5, lane = threadIdx.x & 31;
    int e = blockIdx.x * 8 + wave;
    if (e >= E) return;
    int src = (int)ei[e];
    int dst = (int)ei[(long long)E + e];
    float nrm = dis[src] * dis[dst];
    float4 v = ((const float4*)(h + (long long)src * CH))[lane];   // global_load_b128
    float* op = out + (long long)dst * CH + lane * 4;
    atomicAdd(op + 0, v.x * nrm);
    atomicAdd(op + 1, v.y * nrm);
    atomicAdd(op + 2, v.z * nrm);
    atomicAdd(op + 3, v.w * nrm);
}

// ---------------- zero pool accumulators ----------------
__global__ void k_zero_u32(unsigned* p, int n) {
    int i = blockIdx.x * blockDim.x + threadIdx.x;
    if (i < n) p[i] = 0u;
}

// ---------------- pooling: segment mean-sum + max (post-ReLU, so values >= 0) ----------------
__global__ void k_pool(const long long* __restrict__ batch, const float* __restrict__ h,
                       float* sums, unsigned* maxb, float* counts, int N) {
    int gid = blockIdx.x * blockDim.x + threadIdx.x;
    if (gid >= N * 32) return;
    int n = gid >> 5, q = gid & 31;
    int g = (int)batch[n];
    float4 v4 = ((const float4*)(h + (long long)n * CH))[q];
    float v[4] = {v4.x, v4.y, v4.z, v4.w};
    #pragma unroll
    for (int k = 0; k < 4; ++k) {
        float r = fmaxf(v[k], 0.0f);   // ReLU of layer 3 fused here
        atomicAdd(&sums[g * CH + q * 4 + k], r);
        atomicMax(&maxb[g * CH + q * 4 + k], __float_as_uint(r));  // valid: r >= 0
    }
    if (q == 0) atomicAdd(&counts[g], 1.0f);
}

// ---------------- final MLP: [G,256] @ Wf1 -> relu -> @ Wf2 ----------------
__global__ __launch_bounds__(256)
void k_mlp(const float* __restrict__ sums, const unsigned* __restrict__ maxb,
           const float* __restrict__ counts,
           const float* __restrict__ Wf1, const float* __restrict__ bf1,
           const float* __restrict__ Wf2, const float* __restrict__ bf2,
           float* __restrict__ out) {
    int g = blockIdx.x, t = threadIdx.x;
    __shared__ float pooled[2 * CH];
    __shared__ float hidden[CH];
    if (t < CH) {
        float cnt = fmaxf(counts[g], 1.0f);
        pooled[t] = sums[g * CH + t] / cnt;
    } else {
        pooled[t] = __uint_as_float(maxb[g * CH + (t - CH)]);  // 0 for empty graphs
    }
    __syncthreads();
    if (t < CH) {
        float acc = bf1[t];
        for (int k = 0; k < 2 * CH; ++k) acc += pooled[k] * Wf1[k * CH + t];
        hidden[t] = fmaxf(acc, 0.0f);
    }
    __syncthreads();
    if (t == 0) {
        float acc = bf2[0];
        for (int k = 0; k < CH; ++k) acc += hidden[k] * Wf2[k];
        out[g] = acc;
    }
}

extern "C" void kernel_launch(void* const* d_in, const int* in_sizes, int n_in,
                              void* d_out, int out_size, void* d_ws, size_t ws_size,
                              hipStream_t stream) {
    const float*     x     = (const float*)d_in[0];
    const long long* ei    = (const long long*)d_in[1];   // int64 [2,E] row-major
    const long long* batch = (const long long*)d_in[2];   // int64 [N]
    const float* W1  = (const float*)d_in[3];  const float* b1  = (const float*)d_in[4];
    const float* W2  = (const float*)d_in[5];  const float* b2  = (const float*)d_in[6];
    const float* W3  = (const float*)d_in[7];  const float* b3  = (const float*)d_in[8];
    const float* Wf1 = (const float*)d_in[9];  const float* bf1 = (const float*)d_in[10];
    const float* Wf2 = (const float*)d_in[11]; const float* bf2 = (const float*)d_in[12];
    float* out = (float*)d_out;

    const int N = in_sizes[0] / CH;
    const int E = in_sizes[1] / 2;
    const int G = out_size;               // OUT == 1

    // workspace carving (all 256B aligned)
    char* ws = (char*)d_ws;
    size_t o = 0;
    auto carve = [&](size_t bytes) -> void* {
        void* p = ws + o;
        o += (bytes + 255) & ~(size_t)255;
        return p;
    };
    float* dis  = (float*)carve((size_t)N * 4);
    float* bufA = (float*)carve((size_t)N * CH * 4);
    float* bufB = (float*)carve((size_t)N * CH * 4);
    // sums | maxb | counts laid out contiguously so one zero-kernel covers all
    float*    sums   = (float*)carve((size_t)G * CH * 4 + (size_t)G * CH * 4 + (size_t)G * 4);
    unsigned* maxb   = (unsigned*)(sums + G * CH);
    float*    counts = (float*)(maxb + G * CH);

    const int T = 256;
    const int tiles = (N + RPB - 1) / RPB;
    const int ev = (N * 32 + T - 1) / T;   // elementwise float4 grids

    // degree / symmetric normalization
    k_init_deg<<<(N + T - 1) / T, T, 0, stream>>>(dis, N);
    k_edge_deg<<<(E + T - 1) / T, T, 0, stream>>>(ei, dis, E);
    k_rsqrt  <<<(N + T - 1) / T, T, 0, stream>>>(dis, N);

    // layer 1: x -> bufA (h1), bufB = h1*self+b1, scatter edges into bufB
    k_gemm_wmma   <<<tiles, 256, 0, stream>>>(x, W1, bufA, N, 0);
    k_init_out    <<<ev, T, 0, stream>>>(bufA, dis, b1, bufB, N);
    k_edge_scatter<<<(E + 7) / 8, 256, 0, stream>>>(ei, dis, bufA, bufB, E);

    // layer 2 (ReLU of bufB fused into GEMM A-load)
    k_gemm_wmma   <<<tiles, 256, 0, stream>>>(bufB, W2, bufA, N, 1);
    k_init_out    <<<ev, T, 0, stream>>>(bufA, dis, b2, bufB, N);
    k_edge_scatter<<<(E + 7) / 8, 256, 0, stream>>>(ei, dis, bufA, bufB, E);

    // layer 3
    k_gemm_wmma   <<<tiles, 256, 0, stream>>>(bufB, W3, bufA, N, 1);
    k_init_out    <<<ev, T, 0, stream>>>(bufA, dis, b3, bufB, N);
    k_edge_scatter<<<(E + 7) / 8, 256, 0, stream>>>(ei, dis, bufA, bufB, E);

    // pooling (ReLU of layer 3 fused into pool reads)
    const int zwords = G * CH * 2 + G;
    k_zero_u32<<<(zwords + T - 1) / T, T, 0, stream>>>((unsigned*)sums, zwords);
    k_pool    <<<ev, T, 0, stream>>>(batch, bufB, sums, maxb, counts, N);

    // readout MLP
    k_mlp<<<G, 256, 0, stream>>>(sums, maxb, counts, Wf1, bf1, Wf2, bf2, out);
}
